// QLayer_15788299780327
// MI455X (gfx1250) — matrix operand order repaired and checked by reading, not compiled, not run on hardware
//
#include <hip/hip_runtime.h>
#include <math.h>

// ---- problem constants -----------------------------------------------------
#define BB   64
#define CC   512
#define HW   1024           // H*W = 32*32
#define NN   4
#define DD   128            // C / N
#define KK   1024
#define PTOT 65536          // B*H*W
// d_out layout (floats): [z_q: B*C*H*W][diff: 1][argmin: B*N*H*W][ppl: 1]
#define ZQ_ELEMS   33554432
#define ARGM_ELEMS 262144

typedef __attribute__((ext_vector_type(16))) __bf16 v16bf;
typedef __attribute__((ext_vector_type(8)))  float  v8f;
typedef __attribute__((ext_vector_type(4)))  unsigned int v4u;
typedef __attribute__((ext_vector_type(8)))  int    v8i;
typedef __attribute__((ext_vector_type(4)))  int    v4i;

union Frag { v16bf v; uint4 u[2]; };

// LDS layout for main kernel (dynamic LDS assumed to start at offset 0)
#define ZSTR  520                       // bf16 row stride for z slab (pad, 16B aligned)
#define BSTR  136                       // bf16 row stride for embed chunk (128 + 8 pad)
#define ZBYTES (128 * ZSTR * 2)         // 133120
#define BBYTES (128 * BSTR * 2)         // 34816 (incl. TDM pad rows)
#define BOFF   ZBYTES                   // two B buffers: BOFF, BOFF+BBYTES
#define EOFF   (ZBYTES + 2 * BBYTES)    // two 512B enorm buffers
#define SMEM_MAIN (ZBYTES + 2 * BBYTES + 2 * 512)   // 203776 B (< 320KB WGP LDS)

// ---- TDM: DMA one 128-codes x 128-d bf16 chunk into LDS --------------------
// D# per CDNA5 ISA:
//  g0: [count=1][lds_addr][global_addr lo][global_addr hi | type=2]
//  g1: data_size=2B, pad_enable, pad_interval=256B (code 5), pad_amount=16B (code 3),
//      tensor_dim0=128, tensor_dim1=128, tile_dim0=128, tile_dim1=128,
//      tensor_dim0_stride=128.  Groups 2/3 zero (2-D tile).
__device__ __forceinline__ void tdm_load_chunk(const __bf16* gsrc, unsigned lds_off) {
  unsigned long long ga = (unsigned long long)(uintptr_t)gsrc;
  v4u g0;
  g0[0] = 1u;                                               // count=1, user mode
  g0[1] = lds_off;                                          // lds_addr
  g0[2] = (unsigned)(ga & 0xFFFFFFFFu);                     // global_addr[31:0]
  g0[3] = (unsigned)((ga >> 32) & 0x01FFFFFFu) | (2u << 30);// global_addr[56:32] | type=2
  v8i g1;
  g1[0] = (int)((1u << 16) | (1u << 20) | (5u << 22) | (3u << 25));
  g1[1] = (int)(128u << 16);            // abar_addr=0 | tensor_dim0 lo16
  g1[2] = (int)(128u << 16);            // tensor_dim0 hi16=0 | tensor_dim1 lo16
  g1[3] = (int)(128u << 16);            // tensor_dim1 hi16=0 | tile_dim0=128
  g1[4] = (int)(128u);                  // tile_dim1=128 | tile_dim2=0
  g1[5] = (int)(128u);                  // tensor_dim0_stride lo32
  g1[6] = 0;                            // stride hi16 | dim1_stride lo16
  g1[7] = 0;
  v4i z4 = {0, 0, 0, 0};
  v8i z8 = {0, 0, 0, 0, 0, 0, 0, 0};
  __builtin_amdgcn_tensor_load_to_lds(g0, g1, z4, z4, z8, 0);
}

// ---- K0: embed -> k-major bf16 + ||e||^2 -----------------------------------
__global__ __launch_bounds__(256) void vq_prep(const float* __restrict__ embed,
                                               __bf16* __restrict__ embedT,
                                               float* __restrict__ enorm) {
  int g = blockIdx.x * 256 + threadIdx.x;       // one (n,k) per thread, 4096 total
  if (g >= NN * KK) return;
  int n = g >> 10, k = g & (KK - 1);
  const float* src = embed + (size_t)n * DD * KK + k;   // stride K over d
  __bf16* dst = embedT + (size_t)g * DD;
  float s = 0.f;
#pragma unroll 8
  for (int d = 0; d < DD; ++d) {
    float v = src[(size_t)d * KK];
    s = __builtin_fmaf(v, v, s);
    dst[d] = (__bf16)v;
  }
  enorm[g] = s;
}

// ---- K1: GEMM + argmin (WMMA + TDM double-buffered) -------------------------
__global__ __launch_bounds__(256) void vq_main(const float* __restrict__ x,
                                               const __bf16* __restrict__ embedT,
                                               const float* __restrict__ enorm,
                                               int* __restrict__ idxws,
                                               float* __restrict__ argmin_out) {
  extern __shared__ char smem[];
  __bf16* zbuf = (__bf16*)smem;                     // [128 pts][ZSTR] bf16

  const int tid  = threadIdx.x;
  const int lane = tid & 31;
  const int w    = tid >> 5;                        // wave 0..7 -> 16 points each
  const int l15  = lane & 15;
  const int half = lane >> 4;

  const int p0  = blockIdx.x * 128;                 // 128 points per WG
  const int b   = p0 >> 10;                         // 128 | 1024 so one b per tile
  const int hw0 = p0 & (HW - 1);

  // ---- stage z slab: x[b][c][hw0..hw0+127] -> zbuf[point][c] (bf16) ----
  {
    const int j4 = (tid & 31) * 4;                  // 4 points per thread
    const int c0 = tid >> 5;
    const float* xb = x + (size_t)b * CC * HW + hw0;
    for (int c = c0; c < CC; c += 8) {
      float4 v = *(const float4*)(xb + (size_t)c * HW + j4);
      zbuf[(j4 + 0) * ZSTR + c] = (__bf16)v.x;
      zbuf[(j4 + 1) * ZSTR + c] = (__bf16)v.y;
      zbuf[(j4 + 2) * ZSTR + c] = (__bf16)v.z;
      zbuf[(j4 + 3) * ZSTR + c] = (__bf16)v.w;
    }
  }

  // ---- prologue: async-stage chunk 0 (codes 0..127 of codebook 0) ----
  if (w == 0) tdm_load_chunk(embedT, BOFF);
  if (tid < 128) *(float*)(smem + EOFF + tid * 4) = enorm[tid];

  Frag a[4];
  float best[8]; int bidx[8];

  // 32 uniform chunks: q = n*8 + kc, each 128 codes; double-buffered.
  for (int q = 0; q < 32; ++q) {
    const int n   = q >> 3;
    const int kc  = q & 7;
    const int cur = q & 1;

    if (w == 0) __builtin_amdgcn_s_wait_tensorcnt(0);  // publish chunk q's LDS tile
    __syncthreads();   // chunk q ready; buffer (q+1)&1 (used by chunk q-1) is free

    if (q + 1 < 32) {  // prefetch chunk q+1 while computing chunk q
      if (w == 0)
        tdm_load_chunk(embedT + (size_t)(q + 1) * 128 * DD,
                       BOFF + (unsigned)(((q + 1) & 1) * BBYTES));
      if (tid < 128)
        *(float*)(smem + EOFF + ((q + 1) & 1) * 512 + tid * 4) =
            enorm[(q + 1) * 128 + tid];
    }

    if (kc == 0) {
      // A fragments (16 pts x 128 d) for codebook n, held for 8 chunks.
      // ISA 16-bit A 16x32 layout: lanes<16 hold d{0..7,16..23}, lanes>=16 d{8..15,24..31}.
      const __bf16* zr = zbuf + (w * 16 + l15) * ZSTR + n * DD;
#pragma unroll
      for (int f = 0; f < 4; ++f) {
        a[f].u[0] = *(const uint4*)(zr + f * 32 + half * 8);
        a[f].u[1] = *(const uint4*)(zr + f * 32 + 16 + half * 8);
      }
#pragma unroll
      for (int r = 0; r < 8; ++r) { best[r] = 3.4e38f; bidx[r] = 0; }
    }

    const __bf16* bb = (const __bf16*)(smem + BOFF + cur * BBYTES);
    const float*  eb = (const float*)(smem + EOFF + cur * 512);

#pragma unroll
    for (int kt = 0; kt < 8; ++kt) {
      const int kloc = kt * 16;
      // B fragments. Same K order as A: lanes<16 hold d{0..7,16..23}, lanes>=16
      // d{8..15,24..31} of each 32-deep slice, from the code's row.
      Frag bf[4];
      const __bf16* br = bb + (kloc + l15) * BSTR;
#pragma unroll
      for (int f = 0; f < 4; ++f) {
        bf[f].u[0] = *(const uint4*)(br + f * 32 + half * 8);
        bf[f].u[1] = *(const uint4*)(br + f * 32 + 16 + half * 8);
      }
      v8f acc = {};
#pragma unroll
      for (int f = 0; f < 4; ++f)
        acc = __builtin_amdgcn_wmma_f32_16x16x32_bf16(
            false, a[f].v, false, bf[f].v, (short)0, acc, false, false);

      // dist(p,k) ~ ||e_k||^2 - 2 z.e  (z^2 constant per point, drops from argmin)
      const float en    = eb[kloc + l15];
      const int   kglob = kc * 128 + kloc + l15;
#pragma unroll
      for (int r = 0; r < 8; ++r) {
        float dd = __builtin_fmaf(-2.f, acc[r], en);
        if (dd < best[r]) { best[r] = dd; bidx[r] = kglob; }
      }
    }

    if (kc == 7) {
      // reduce across the 16 lanes sharing each C row (lanes 0-15: rows 0-7, 16-31: 8-15)
#pragma unroll
      for (int r = 0; r < 8; ++r) {
        for (int off = 1; off < 16; off <<= 1) {
          float ob = __shfl_xor(best[r], off);
          int   oi = __shfl_xor(bidx[r], off);
          if (ob < best[r] || (ob == best[r] && oi < bidx[r])) { best[r] = ob; bidx[r] = oi; }
        }
      }
      if (l15 == 0) {
        const int rowbase = w * 16 + half * 8;
#pragma unroll
        for (int r = 0; r < 8; ++r) {
          const int p = p0 + rowbase + r;
          idxws[n * PTOT + p] = bidx[r];
          argmin_out[((size_t)b * NN + n) * HW + (hw0 + rowbase + r)] = (float)bidx[r];
        }
      }
    }
  }
}

// ---- K2: z_q = x + (q - x), coalesced in [C][HW]; MSE partials per block ----
__global__ __launch_bounds__(256) void vq_out(const float* __restrict__ x,
                                              const float* __restrict__ embed,
                                              const int* __restrict__ idxws,
                                              float* __restrict__ zq,
                                              float* __restrict__ diffpart) {
  __shared__ float sred[256];
  const int tid = threadIdx.x;
  const int bx  = blockIdx.x;            // B*N*D = 32768 blocks
  const int d   = bx & (DD - 1);
  const int n   = (bx >> 7) & (NN - 1);
  const int b   = bx >> 9;
  const int c   = n * DD + d;
  const int hw  = tid * 4;

  const int4  i4 = *(const int4*)(idxws + (size_t)n * PTOT + b * HW + hw);
  const float* erow = embed + ((size_t)n * DD + d) * KK;
  const float4 x4 = *(const float4*)(x + ((size_t)b * CC + c) * HW + hw);
  float q0 = erow[i4.x], q1 = erow[i4.y], q2 = erow[i4.z], q3 = erow[i4.w];
  float4 z4;
  z4.x = x4.x + (q0 - x4.x);
  z4.y = x4.y + (q1 - x4.y);
  z4.z = x4.z + (q2 - x4.z);
  z4.w = x4.w + (q3 - x4.w);
  *(float4*)(zq + ((size_t)b * CC + c) * HW + hw) = z4;

  float s = (q0 - x4.x) * (q0 - x4.x) + (q1 - x4.y) * (q1 - x4.y) +
            (q2 - x4.z) * (q2 - x4.z) + (q3 - x4.w) * (q3 - x4.w);
  sred[tid] = s;
  __syncthreads();
  for (int o = 128; o > 0; o >>= 1) {
    if (tid < o) sred[tid] += sred[tid + o];
    __syncthreads();
  }
  if (tid == 0) diffpart[bx] = sred[0];
}

// ---- K3: finalize diff + histogram -> perplexity ----------------------------
__global__ __launch_bounds__(256) void vq_finalize(const int* __restrict__ idxws,
                                                   const float* __restrict__ diffpart,
                                                   float* __restrict__ diff_out,
                                                   float* __restrict__ ppl_out) {
  __shared__ float shist[NN * KK];
  __shared__ float sred[256];
  const int tid = threadIdx.x;

  // diff = sum(partials) / (B*H*W*C)
  float s = 0.f;
  for (int i = tid; i < BB * NN * DD; i += 256) s += diffpart[i];
  sred[tid] = s;
  __syncthreads();
  for (int o = 128; o > 0; o >>= 1) {
    if (tid < o) sred[tid] += sred[tid + o];
    __syncthreads();
  }
  if (tid == 0) *diff_out = sred[0] / (float)ZQ_ELEMS;
  __syncthreads();

  // histogram over codes (LDS ds-atomics)
  for (int i = tid; i < NN * KK; i += 256) shist[i] = 0.f;
  __syncthreads();
  for (int i = tid; i < NN * PTOT; i += 256) {
    int idx = idxws[i];
    int n = i >> 16;                      // PTOT = 65536
    atomicAdd(&shist[n * KK + idx], 1.0f);
  }
  __syncthreads();

  // ppl = mean_n exp(-sum_k p*log(p+1e-10))
  float acc = 0.f;
  for (int n = 0; n < NN; ++n) {
    float h = 0.f;
    for (int k = tid; k < KK; k += 256) {
      float pr = shist[n * KK + k] * (1.0f / (float)PTOT);
      h += pr * logf(pr + 1e-10f);
    }
    sred[tid] = h;
    __syncthreads();
    for (int o = 128; o > 0; o >>= 1) {
      if (tid < o) sred[tid] += sred[tid + o];
      __syncthreads();
    }
    if (tid == 0) acc += expf(-sred[0]);
    __syncthreads();
  }
  if (tid == 0) *ppl_out = acc * (1.0f / (float)NN);
}

// ---- launcher ---------------------------------------------------------------
extern "C" void kernel_launch(void* const* d_in, const int* in_sizes, int n_in,
                              void* d_out, int out_size, void* d_ws, size_t ws_size,
                              hipStream_t stream) {
  const float* x     = (const float*)d_in[0];
  const float* embed = (const float*)d_in[1];

  float* out     = (float*)d_out;
  float* zq      = out;
  float* diff    = out + ZQ_ELEMS;
  float* argmin  = out + ZQ_ELEMS + 1;
  float* ppl     = out + ZQ_ELEMS + 1 + ARGM_ELEMS;

  char* ws = (char*)d_ws;
  __bf16* embedT  = (__bf16*)ws;                                   // 1,048,576 B
  float*  enorm   = (float*)(ws + 1048576);                        //    16,384 B
  int*    idxws   = (int*)(ws + 1048576 + 16384);                  // 1,048,576 B
  float*  diffp   = (float*)(ws + 1048576 + 16384 + 1048576);      //   131,072 B

  vq_prep<<<(NN * KK + 255) / 256, 256, 0, stream>>>(embed, embedT, enorm);
  vq_main<<<PTOT / 128, 256, SMEM_MAIN, stream>>>(x, embedT, enorm, idxws, argmin);
  vq_out<<<BB * NN * DD, 256, 0, stream>>>(x, embed, idxws, zq, diffp);
  vq_finalize<<<1, 256, 0, stream>>>(idxws, diffp, diff, ppl);
}
